// MultiAttn_29669634080806
// MI455X (gfx1250) — matrix operand order repaired and checked by reading, not compiled, not run on hardware
//
#include <hip/hip_runtime.h>
#include <hip/hip_bf16.h>

#define DIM   768
#define HEADS 4
#define DHEAD 192
#define NUSR  4096
#define NBRS  32
#define TPB   512          // 16 waves of 32
#define XP    (DIM + 8)    // padded LDS row stride (ushorts) to avoid bank conflicts

typedef __attribute__((ext_vector_type(16))) __bf16         v16bf;
typedef __attribute__((ext_vector_type(16))) unsigned short v16u;
typedef __attribute__((ext_vector_type(8)))  float          v8f;

__device__ __forceinline__ unsigned short f2bf(float f) {
    union { float f; unsigned u; } c; c.f = f;
    unsigned u = c.u + 0x7FFFu + ((c.u >> 16) & 1u);   // round-to-nearest-even
    return (unsigned short)(u >> 16);
}
__device__ __forceinline__ float bf2f(unsigned short h) {
    union { unsigned u; float f; } c; c.u = ((unsigned)h) << 16;
    return c.f;
}

// Copy features to the evolving table and convert the 4 weight matrices to bf16
// (row-major, same layout as the fp32 originals).
__global__ void MultiAttn_prep_kernel(const float* __restrict__ feats_in,
                                      float* __restrict__ feats_out,
                                      const float* __restrict__ in_proj_w,
                                      const float* __restrict__ out_proj_w,
                                      unsigned short* __restrict__ wq,
                                      unsigned short* __restrict__ wk,
                                      unsigned short* __restrict__ wv,
                                      unsigned short* __restrict__ wo) {
    long i = (long)blockIdx.x * blockDim.x + threadIdx.x;
    const long WD = (long)DIM * DIM;
    if (i < (long)NUSR * DIM) feats_out[i] = feats_in[i];
    if (i < WD) {
        wq[i] = f2bf(in_proj_w[i]);            // Wq rows 0..D-1
        wk[i] = f2bf(in_proj_w[WD + i]);       // Wk rows D..2D-1
        wv[i] = f2bf(in_proj_w[2 * WD + i]);   // Wv rows 2D..3D-1
        wo[i] = f2bf(out_proj_w[i]);
    }
}

// Single persistent workgroup: serial scan over the 4096 users.
__global__ __launch_bounds__(TPB)
void MultiAttn_seq_kernel(const int* __restrict__ nbr,
                          float* __restrict__ feats,
                          const unsigned short* __restrict__ wq,
                          const unsigned short* __restrict__ wk,
                          const unsigned short* __restrict__ wv,
                          const unsigned short* __restrict__ wo,
                          const float* __restrict__ in_b,
                          const float* __restrict__ out_b) {
    extern __shared__ unsigned char smem[];
    unsigned short* x_bf = (unsigned short*)smem;        // [NBRS][XP] gathered x, bf16
    float* m_s  = (float*)(x_bf + NBRS * XP);            // [HEADS][DIM]  m_h = Wk_h^T q_h
    float* q_s  = m_s + HEADS * DIM;                     // [DIM]
    float* o_s  = q_s + DIM;                             // [DIM] attn-weighted V
    float* sc_s = o_s + DIM;                             // [HEADS][NBRS] scores
    float* at_s = sc_s + HEADS * NBRS;                   // [HEADS][NBRS] attn
    float* cb_s = at_s + HEADS * NBRS;                   // [HEADS] q·bk per head
    int*   id_s = (int*)(cb_s + HEADS);                  // [NBRS]

    const int tid  = threadIdx.x;
    const int lane = tid & 31;
    const int wave = tid >> 5;
    const int hi   = (lane >> 4) & 1;  // half-wave
    const int lm   = lane & 15;
    const float scale = 0.0721687836487032f;  // 1/sqrt(192)

    const float* bq = in_b;
    const float* bk = in_b + DIM;
    const float* bv = in_b + 2 * DIM;

    for (int t = 0; t < NUSR; ++t) {
        // ---- indices ----
        if (tid < NBRS) id_s[tid] = nbr[t * NBRS + tid];
        __syncthreads();

        // ---- gather neighbor rows -> bf16 LDS (volatile: rows may have been
        //      rewritten by earlier steps of this same workgroup) ----
        for (int i = tid; i < NBRS * DIM; i += TPB) {
            int kk = i / DIM, d = i % DIM;
            volatile const float* src = feats + (long)id_s[kk] * DIM + d;
            x_bf[kk * XP + d] = f2bf(*src);
        }
        __syncthreads();

        // ---- q[n] = bq[n] + sum_c x0[c] * Wq[n][c] ----
        for (int n = tid; n < DIM; n += TPB) {
            const unsigned short* wr = wq + (long)n * DIM;
            float acc = bq[n];
            for (int c = 0; c < DIM; ++c) acc += bf2f(x_bf[c]) * bf2f(wr[c]);
            q_s[n] = acc;
        }
        __syncthreads();

        // ---- per-head constant  cb[h] = q_h · bk_h ----
        if (tid < HEADS) {
            float acc = 0.f;
            for (int d = 0; d < DHEAD; ++d)
                acc += q_s[tid * DHEAD + d] * bk[tid * DHEAD + d];
            cb_s[tid] = acc;
        }
        // ---- m_h[c] = sum_{j in head h} Wk[j][c] * q[j]  (fold Wk through q) ----
        for (int i = tid; i < DIM * HEADS; i += TPB) {
            int c = i % DIM, h = i / DIM;
            int j0 = h * DHEAD;
            float acc = 0.f;
            for (int j = 0; j < DHEAD; ++j)
                acc += bf2f(wk[(long)(j0 + j) * DIM + c]) * q_s[j0 + j];
            m_s[h * DIM + c] = acc;
        }
        __syncthreads();

        // ---- scores s[h][kk] = scale * (x[kk]·m_h + cb[h]) ----
        if (tid < HEADS * NBRS) {
            int kk = tid % NBRS, h = tid / NBRS;
            const float* mh = m_s + h * DIM;
            const unsigned short* xr = x_bf + kk * XP;
            float acc = cb_s[h];
            for (int c = 0; c < DIM; ++c) acc += bf2f(xr[c]) * mh[c];
            sc_s[h * NBRS + kk] = acc * scale;
        }
        __syncthreads();

        // ---- softmax per head ----
        if (tid < HEADS) {
            float mx = -1e30f;
            for (int kk = 0; kk < NBRS; ++kk) mx = fmaxf(mx, sc_s[tid * NBRS + kk]);
            float sum = 0.f;
            for (int kk = 0; kk < NBRS; ++kk) {
                float e = __expf(sc_s[tid * NBRS + kk] - mx);
                at_s[tid * NBRS + kk] = e; sum += e;
            }
            float inv = 1.f / sum;
            for (int kk = 0; kk < NBRS; ++kk) at_s[tid * NBRS + kk] *= inv;
        }
        __syncthreads();

        // ---- V = X @ Wv^T via WMMA bf16; attention weights applied directly
        //      to the accumulator rows, so V is never materialized.
        //      48 N-tiles of 16 cols; 16 waves -> 3 N-tiles each.
        //      B fragment (Wv) loaded ONCE per k-step and shared by both M-tiles
        //      (two accumulators) -> halves the dominant L2 weight stream.
        for (int it = 0; it < 3; ++it) {
            int nt  = wave * 3 + it;
            int n_g = nt * 16 + lm;          // this lane's output column
            int h   = n_g / DHEAD;           // head of that column
            v8f acc0 = {};                   // rows 0..15  (kk tile 0)
            v8f acc1 = {};                   // rows 16..31 (kk tile 1)
            for (int ks = 0; ks < 24; ++ks) {
                // B fragment (32x16 bf16): B(k,n) = Wv[n][k] -> two runs of 8 elems (k = 8*hi.., 16+8*hi..)
                const unsigned short* br = wv + (long)n_g * DIM + ks * 32 + hi * 8;
                v16u bu;
                #pragma unroll
                for (int e = 0; e < 16; ++e) bu[e] = br[e < 8 ? e : e + 8];
                v16bf b = __builtin_bit_cast(v16bf, bu);
                // A fragments (16x32 bf16) for both M-tiles: row = mt*16+lm
                const unsigned short* a0 = x_bf + lm * XP        + ks * 32 + (hi ? 8 : 0);
                const unsigned short* a1 = x_bf + (16 + lm) * XP + ks * 32 + (hi ? 8 : 0);
                v16u au0, au1;
                #pragma unroll
                for (int e = 0; e < 8; ++e) {
                    au0[e] = a0[e]; au0[8 + e] = a0[16 + e];
                    au1[e] = a1[e]; au1[8 + e] = a1[16 + e];
                }
                acc0 = __builtin_amdgcn_wmma_f32_16x16x32_bf16(
                    false, __builtin_bit_cast(v16bf, au0), false, b,
                    (short)0, acc0, false, false);
                acc1 = __builtin_amdgcn_wmma_f32_16x16x32_bf16(
                    false, __builtin_bit_cast(v16bf, au1), false, b,
                    (short)0, acc1, false, false);
            }
            // weight accumulator rows by attn and fold into per-column sum
            float osum = 0.f;
            #pragma unroll
            for (int r = 0; r < 8; ++r) {
                int row = r + 8 * hi;                    // D layout: M = r + 8*hi
                osum += at_s[h * NBRS + row]      * acc0[r];
                osum += at_s[h * NBRS + 16 + row] * acc1[r];
            }
            osum += __shfl_xor(osum, 16, 32);            // combine half-waves (same col)
            if (hi == 0) o_s[n_g] = osum + bv[n_g];      // sum(attn)=1 -> +bv once
        }
        __syncthreads();

        // ---- out projection + scatter:  out[j] = bo[j] + sum_n o[n]*Wo[j][n] ----
        for (int j = tid; j < DIM; j += TPB) {
            const unsigned short* wr = wo + (long)j * DIM;
            float acc = out_b[j];
            for (int n = 0; n < DIM; ++n) acc += o_s[n] * bf2f(wr[n]);
            feats[(long)t * DIM + j] = acc;
        }
        __threadfence();       // make the scatter visible to next step's gather
        __syncthreads();
    }
}

extern "C" void kernel_launch(void* const* d_in, const int* in_sizes, int n_in,
                              void* d_out, int out_size, void* d_ws, size_t ws_size,
                              hipStream_t stream) {
    (void)in_sizes; (void)n_in; (void)out_size; (void)ws_size;
    const int*   nbr      = (const int*)d_in[0];
    const float* feats_in = (const float*)d_in[1];
    const float* in_w     = (const float*)d_in[2];
    const float* in_b     = (const float*)d_in[3];
    const float* out_w    = (const float*)d_in[4];
    const float* out_b    = (const float*)d_in[5];
    float* feats = (float*)d_out;

    const size_t WD = (size_t)DIM * DIM;
    unsigned short* wq = (unsigned short*)d_ws;
    unsigned short* wk = wq + WD;
    unsigned short* wv = wk + WD;
    unsigned short* wo = wv + WD;

    long total = (long)NUSR * DIM;           // covers both copy and weight convert
    int blocks = (int)((total + 255) / 256);
    MultiAttn_prep_kernel<<<blocks, 256, 0, stream>>>(feats_in, feats, in_w, out_w,
                                                      wq, wk, wv, wo);

    size_t smem = (size_t)NBRS * XP * sizeof(unsigned short)
                + (size_t)(HEADS * DIM + DIM + DIM + 2 * HEADS * NBRS + HEADS) * sizeof(float)
                + (size_t)NBRS * sizeof(int);
    MultiAttn_seq_kernel<<<1, TPB, smem, stream>>>(nbr, feats, wq, wk, wv, wo,
                                                   in_b, out_b);
}
